// Swin_2D_v1_24300924961186
// MI455X (gfx1250) — matrix operand order repaired and checked by reading, not compiled, not run on hardware
//
#include <hip/hip_runtime.h>
#include <math.h>

// ---------------- CDNA5 WMMA types ----------------
typedef __attribute__((ext_vector_type(16))) _Float16 v16h;
typedef __attribute__((ext_vector_type(8)))  float    v8f;

#define B_    32
#define H_    56
#define W_    56
#define C_    128
#define WSZ   7
#define SS_   3
#define N_    49          // tokens per window (padded to 64)
#define NH_   4
#define HD_   32
#define NWIN  2048        // B_ * 64 windows
#define SCALE_ 0.17677669529663687f   // 1/sqrt(32)

// ---------------- WMMA fragment helpers (contiguous k = 16*(lane/16)+s, same in A and B) ----------------
// A 16x32 f16: lane<16 -> row m0+lane, K = {k0..k0+15}
//              lane>=16 -> row m0+lane-16, K = {k0+16..k0+31}
__device__ __forceinline__ v16h load_a_frag(const _Float16* base, int m0, int k0,
                                            int ld, int lane) {
  const _Float16* p = base + (size_t)(m0 + (lane & 15)) * ld + k0 + ((lane >> 4) << 4);
  v16h a;
#pragma unroll
  for (int h = 0; h < 8; ++h) a[h] = p[h];
#pragma unroll
  for (int h = 0; h < 8; ++h) a[8 + h] = p[8 + h];
  return a;
}

// B 32x16 f16 from column-major storage Bt[n][k] (leading dim = K extent):
// lane holds column n = n0+(lane&15), halves = K k0+16*(lane>>4)+h  (contiguous)
__device__ __forceinline__ v16h load_b_ct(const _Float16* bt, int n0, int k0,
                                          int ld, int lane) {
  const _Float16* p = bt + (size_t)(n0 + (lane & 15)) * ld + k0 + ((lane >> 4) << 4);
  v16h b;
#pragma unroll
  for (int h = 0; h < 16; ++h) b[h] = p[h];
  return b;
}

__device__ __forceinline__ v8f wmma_f16(v16h a, v16h b, v8f c) {
  return __builtin_amdgcn_wmma_f32_16x16x32_f16(false, a, false, b, (short)0, c,
                                                false, false);
}

// Swin shift-region id for a (shifted-canvas) coordinate
__device__ __forceinline__ int regio(int p) {
  return (p < H_ - WSZ) ? 0 : (p < H_ - SS_) ? 1 : 2;
}

// gfx1250 async global->LDS copy (ASYNCcnt-tracked, ISA §15.18 op 98)
__device__ __forceinline__ void async_copy_b128(unsigned lds_off,
                                                const void* gaddr) {
  asm volatile("global_load_async_to_lds_b128 %0, %1, off"
               :: "v"(lds_off), "v"(gaddr) : "memory");
}
__device__ __forceinline__ void wait_asynccnt0() {
  asm volatile("s_wait_asynccnt 0x0" ::: "memory");
}

// ---------------- Kernel W: fp32 weights -> transposed f16 ----------------
__global__ __launch_bounds__(256)
void swin_wconv(const float* __restrict__ qkv_w, const float* __restrict__ proj_w,
                const float* __restrict__ fc1_w, const float* __restrict__ fc2_w,
                _Float16* __restrict__ wq_t, _Float16* __restrict__ pj_t,
                _Float16* __restrict__ f1_t, _Float16* __restrict__ f2_t) {
  int idx = blockIdx.x * 256 + threadIdx.x;            // 0 .. 65535
  if (idx < 49152) {                                   // qkv [128][384] -> [384][128]
    int n = idx / 128, k = idx % 128;
    wq_t[idx] = (_Float16)qkv_w[k * 384 + n];
  }
  if (idx < 16384) {                                   // proj [128][128] -> [128][128]^T
    int n = idx / 128, k = idx % 128;
    pj_t[idx] = (_Float16)proj_w[k * 128 + n];
  }
  if (idx < 65536) {                                   // fc1 [128][512] -> [512][128]
    int n = idx / 128, k = idx % 128;
    f1_t[idx] = (_Float16)fc1_w[k * 512 + n];
  }
  if (idx < 65536) {                                   // fc2 [512][128] -> [128][512]
    int n = idx / 512, k = idx % 512;
    f2_t[idx] = (_Float16)fc2_w[k * 128 + n];
  }
}

// ---------------- Kernel A: fused window attention ----------------
// One workgroup (8 wave32) per window. LDS ~112 KB of the 320 KB WGP pool.
__global__ __launch_bounds__(256)
void swin_attn(const float* __restrict__ x,
               const float* __restrict__ n1w, const float* __restrict__ n1b,
               const _Float16* __restrict__ wq_t, const float* __restrict__ qkv_b,
               const _Float16* __restrict__ pj_t, const float* __restrict__ proj_b,
               const float* __restrict__ rel_tab,
               float* __restrict__ x1) {
  __shared__ _Float16 s_x[64 * 128];        // LN1'd shifted window tokens (f16)
  __shared__ _Float16 s_qkv[64 * 256];      // Q(scaled)|K per token
  __shared__ _Float16 s_vt[128 * 64];       // V transposed: [head*32+d][token]
  __shared__ _Float16 s_attn[4 * 64 * 64];  // softmaxed attention per head
  __shared__ _Float16 s_o[64 * 128];        // head-concat attn@V
  __shared__ short    s_ii[64], s_jj[64], s_cc[64];  // token geometry

  const int wgid = blockIdx.x;
  const int b  = wgid >> 6;
  const int wy = (wgid >> 3) & 7;
  const int wx = wgid & 7;
  const int tid  = threadIdx.x;
  const int wave = tid >> 5;
  const int lane = tid & 31;

  // fire prefetch of the QKV weight panel while LN runs (global_prefetch_b8)
  for (int off = tid * 128; off < 384 * 128 * 2; off += 256 * 128)
    __builtin_prefetch((const char*)wq_t + off, 0, 0);

  // token geometry tables
  if (tid < 64) {
    if (tid < N_) {
      const int i = tid / 7, j = tid % 7;
      s_ii[tid] = (short)i;
      s_jj[tid] = (short)j;
      s_cc[tid] = (short)(regio(wy * 7 + i) * 3 + regio(wx * 7 + j));
    } else {
      s_ii[tid] = 0; s_jj[tid] = 0; s_cc[tid] = 0;
    }
  }

  // ---- Stage 1: LN1 + cyclic shift + window gather -> s_x (rows>=49 zero)
  for (int row = wave; row < 64; row += 8) {
    if (row < N_) {
      const int i = row / 7, j = row % 7;
      const int sr = (wy * 7 + i + SS_) % H_;
      const int sc = (wx * 7 + j + SS_) % W_;
      const float* xr = x + ((size_t)b * (H_ * W_) + sr * W_ + sc) * C_;
      const float4 vv = ((const float4*)xr)[lane];       // channels 4*lane..+3
      float v0[4] = {vv.x, vv.y, vv.z, vv.w};
      float s = v0[0] + v0[1] + v0[2] + v0[3];
      for (int mm = 16; mm >= 1; mm >>= 1) s += __shfl_xor(s, mm, 32);
      const float mu = s * (1.f / 128.f);
      float vs = 0.f;
#pragma unroll
      for (int u = 0; u < 4; ++u) { float d = v0[u] - mu; vs += d * d; }
      for (int mm = 16; mm >= 1; mm >>= 1) vs += __shfl_xor(vs, mm, 32);
      const float rstd = rsqrtf(vs * (1.f / 128.f) + 1e-5f);
#pragma unroll
      for (int u = 0; u < 4; ++u) {
        const int c = lane * 4 + u;
        s_x[row * 128 + c] = (_Float16)((v0[u] - mu) * rstd * n1w[c] + n1b[c]);
      }
    } else {
#pragma unroll
      for (int u = 0; u < 4; ++u) s_x[row * 128 + lane * 4 + u] = (_Float16)0.f;
    }
  }
  __syncthreads();

  // ---- Stage 2: QKV GEMM [64x128]@[128x384]; Q pre-scaled; V stored transposed
  for (int t = wave; t < 96; t += 8) {
    const int m0 = (t & 3) * 16;
    const int n0 = (t >> 2) * 16;
    v8f acc = {0, 0, 0, 0, 0, 0, 0, 0};
    for (int k0 = 0; k0 < 128; k0 += 32) {
      v16h a  = load_a_frag(s_x, m0, k0, 128, lane);
      v16h bf = load_b_ct(wq_t, n0, k0, 128, lane);
      acc = wmma_f16(a, bf, acc);
    }
    const int n = n0 + (lane & 15);
    const float bias = qkv_b[n];
    const int rbase = m0 + ((lane >> 4) << 3);
    if (n0 < 256) {                       // Q | K : row-major [64][256]
      const float sc = (n < 128) ? SCALE_ : 1.f;
#pragma unroll
      for (int r = 0; r < 8; ++r)
        s_qkv[(rbase + r) * 256 + n] = (_Float16)((acc[r] + bias) * sc);
    } else {                              // V : transposed [dim][token], contiguous
      const int d = n - 256;
#pragma unroll
      for (int r = 0; r < 8; ++r)
        s_vt[d * 64 + rbase + r] = (_Float16)(acc[r] + bias);
    }
  }
  __syncthreads();

  // ---- Stage 3: QK^T (+rel bias +shift mask) + softmax -> s_attn
  // 16 strips = 4 heads x 4 row-tiles; each wave does 2 strips.
  for (int s = wave; s < 16; s += 8) {
    const int head = s >> 2;
    const int m0 = (s & 3) * 16;
    const int nsub = lane & 15;
    const int grp  = lane >> 4;
    v16h aQ = load_a_frag(s_qkv, m0, head * 32, 256, lane);  // K-dim == HD == 32
    v8f sc4[4];
#pragma unroll
    for (int ct = 0; ct < 4; ++ct) {
      v8f acc = {0, 0, 0, 0, 0, 0, 0, 0};
      v16h bK = load_b_ct(s_qkv + 128 + head * 32, ct * 16, 0, 256, lane);
      sc4[ct] = wmma_f16(aQ, bK, acc);
    }
    float rowmax[8], rowsum[8];
#pragma unroll
    for (int r = 0; r < 8; ++r) rowmax[r] = -3.0e38f;
    for (int ct = 0; ct < 4; ++ct) {
      const int mcol = ct * 16 + nsub;
#pragma unroll
      for (int r = 0; r < 8; ++r) {
        const int nrow = m0 + grp * 8 + r;
        float v = sc4[ct][r];
        if (mcol < N_ && nrow < N_) {
          const int idx = (s_ii[nrow] - s_ii[mcol] + 6) * 13 +
                          (s_jj[nrow] - s_jj[mcol] + 6);
          v += rel_tab[idx * NH_ + head];
          if (s_cc[nrow] != s_cc[mcol]) v += -100.f;
        } else if (mcol >= N_) {
          v = -1.0e4f;                       // exclude padded keys
        }
        sc4[ct][r] = v;
        rowmax[r] = fmaxf(rowmax[r], v);
      }
    }
#pragma unroll
    for (int r = 0; r < 8; ++r)
      for (int mm = 8; mm >= 1; mm >>= 1)
        rowmax[r] = fmaxf(rowmax[r], __shfl_xor(rowmax[r], mm, 32));
#pragma unroll
    for (int r = 0; r < 8; ++r) rowsum[r] = 0.f;
    for (int ct = 0; ct < 4; ++ct)
#pragma unroll
      for (int r = 0; r < 8; ++r) {
        const float e = __expf(sc4[ct][r] - rowmax[r]);
        sc4[ct][r] = e;
        rowsum[r] += e;
      }
#pragma unroll
    for (int r = 0; r < 8; ++r) {
      for (int mm = 8; mm >= 1; mm >>= 1) rowsum[r] += __shfl_xor(rowsum[r], mm, 32);
      rowsum[r] = 1.f / rowsum[r];
    }
    for (int ct = 0; ct < 4; ++ct)
#pragma unroll
      for (int r = 0; r < 8; ++r)
        s_attn[head * 4096 + (m0 + grp * 8 + r) * 64 + ct * 16 + nsub] =
            (_Float16)(sc4[ct][r] * rowsum[r]);
  }
  __syncthreads();

  // ---- Stage 4: attn @ V -> s_o   (V^T gives contiguous B fragments)
  for (int t = wave; t < 32; t += 8) {
    const int head = t >> 3;
    const int m0 = ((t >> 1) & 3) * 16;
    const int n0 = (t & 1) * 16;
    v8f acc = {0, 0, 0, 0, 0, 0, 0, 0};
    for (int k0 = 0; k0 < 64; k0 += 32) {
      v16h a  = load_a_frag(s_attn + head * 4096, m0, k0, 64, lane);
      v16h bf = load_b_ct(s_vt + head * 32 * 64, n0, k0, 64, lane);
      acc = wmma_f16(a, bf, acc);
    }
    const int n = n0 + (lane & 15);
    const int rbase = m0 + ((lane >> 4) << 3);
#pragma unroll
    for (int r = 0; r < 8; ++r)
      s_o[(rbase + r) * 128 + head * 32 + n] = (_Float16)acc[r];
  }
  __syncthreads();

  // ---- Stage 5: proj GEMM + window-reverse + reverse-shift + residual -> x1
  for (int t = wave; t < 32; t += 8) {
    const int m0 = (t & 3) * 16;
    const int n0 = (t >> 2) * 16;
    v8f acc = {0, 0, 0, 0, 0, 0, 0, 0};
    for (int k0 = 0; k0 < 128; k0 += 32) {
      v16h a  = load_a_frag(s_o, m0, k0, 128, lane);
      v16h bf = load_b_ct(pj_t, n0, k0, 128, lane);
      acc = wmma_f16(a, bf, acc);
    }
    const int n = n0 + (lane & 15);
    const float pb = proj_b[n];
    const int rbase = m0 + ((lane >> 4) << 3);
#pragma unroll
    for (int r = 0; r < 8; ++r) {
      const int row = rbase + r;
      if (row < N_) {
        const int i = row / 7, j = row % 7;
        const int sr = (wy * 7 + i + SS_) % H_;
        const int sc = (wx * 7 + j + SS_) % W_;
        const size_t off = ((size_t)b * (H_ * W_) + sr * W_ + sc) * C_ + n;
        x1[off] = x[off] + acc[r] + pb;
      }
    }
  }
}

// ---------------- Kernel B: LN2 -> f16 ----------------
__global__ __launch_bounds__(256)
void swin_ln2(const float* __restrict__ x1, const float* __restrict__ w,
              const float* __restrict__ bta, _Float16* __restrict__ hout) {
  const int row  = blockIdx.x * 8 + (threadIdx.x >> 5);
  const int lane = threadIdx.x & 31;
  const float* xr = x1 + (size_t)row * 128;
  const float4 vv = ((const float4*)xr)[lane];
  float v0[4] = {vv.x, vv.y, vv.z, vv.w};
  float s = v0[0] + v0[1] + v0[2] + v0[3];
  for (int mm = 16; mm >= 1; mm >>= 1) s += __shfl_xor(s, mm, 32);
  const float mu = s * (1.f / 128.f);
  float vs = 0.f;
#pragma unroll
  for (int u = 0; u < 4; ++u) { float d = v0[u] - mu; vs += d * d; }
  for (int mm = 16; mm >= 1; mm >>= 1) vs += __shfl_xor(vs, mm, 32);
  const float rstd = rsqrtf(vs * (1.f / 128.f) + 1e-5f);
#pragma unroll
  for (int u = 0; u < 4; ++u) {
    const int c = lane * 4 + u;
    hout[(size_t)row * 128 + c] = (_Float16)((v0[u] - mu) * rstd * w[c] + bta[c]);
  }
}

// ---------------- Kernel C: fused MLP (FC1 + exact GELU + FC2 + residual) ---
__global__ __launch_bounds__(256)
void swin_mlp(const float* __restrict__ x1, const _Float16* __restrict__ hin,
              const _Float16* __restrict__ f1_t, const float* __restrict__ f1_b,
              const _Float16* __restrict__ f2_t, const float* __restrict__ f2_b,
              float* __restrict__ out) {
  __shared__ _Float16 s_h[64 * 128];
  __shared__ _Float16 s_m[64 * 512];
  const int row0 = blockIdx.x * 64;
  const int tid  = threadIdx.x;
  const int wave = tid >> 5;
  const int lane = tid & 31;

  // async global->LDS fill of the 64 LN2'd token rows (ASYNCcnt path)
  {
    const _Float16* gsrc = hin + (size_t)row0 * 128;
    for (int idx = tid; idx < (64 * 128) / 8; idx += 256) {
      const unsigned lds_off = (unsigned)(uintptr_t)(s_h + idx * 8);
      async_copy_b128(lds_off, gsrc + idx * 8);
    }
    wait_asynccnt0();
  }
  __syncthreads();

  // FC1 [64x128]@[128x512] + bias + GELU -> s_m (f16)
  for (int t = wave; t < 128; t += 8) {
    const int m0 = (t & 3) * 16;
    const int n0 = (t >> 2) * 16;
    v8f acc = {0, 0, 0, 0, 0, 0, 0, 0};
    for (int k0 = 0; k0 < 128; k0 += 32) {
      v16h a  = load_a_frag(s_h, m0, k0, 128, lane);
      v16h bf = load_b_ct(f1_t, n0, k0, 128, lane);
      acc = wmma_f16(a, bf, acc);
    }
    const int n = n0 + (lane & 15);
    const float bb = f1_b[n];
    const int rbase = m0 + ((lane >> 4) << 3);
#pragma unroll
    for (int r = 0; r < 8; ++r) {
      float v = acc[r] + bb;
      v = 0.5f * v * (1.f + erff(v * 0.70710678118654752f));
      s_m[(rbase + r) * 512 + n] = (_Float16)v;
    }
  }
  __syncthreads();

  // FC2 [64x512]@[512x128] + bias + residual -> out
  for (int t = wave; t < 32; t += 8) {
    const int m0 = (t & 3) * 16;
    const int n0 = (t >> 2) * 16;
    v8f acc = {0, 0, 0, 0, 0, 0, 0, 0};
    for (int k0 = 0; k0 < 512; k0 += 32) {
      v16h a  = load_a_frag(s_m, m0, k0, 512, lane);
      v16h bf = load_b_ct(f2_t, n0, k0, 512, lane);
      acc = wmma_f16(a, bf, acc);
    }
    const int n = n0 + (lane & 15);
    const float bb = f2_b[n];
    const int rbase = m0 + ((lane >> 4) << 3);
#pragma unroll
    for (int r = 0; r < 8; ++r) {
      const size_t off = (size_t)(row0 + rbase + r) * 128 + n;
      out[off] = x1[off] + acc[r] + bb;
    }
  }
}

// ---------------- Launch ----------------
extern "C" void kernel_launch(void* const* d_in, const int* in_sizes, int n_in,
                              void* d_out, int out_size, void* d_ws, size_t ws_size,
                              hipStream_t stream) {
  (void)in_sizes; (void)n_in; (void)out_size; (void)ws_size;
  const float* x       = (const float*)d_in[0];
  const float* n1w     = (const float*)d_in[1];
  const float* n1b     = (const float*)d_in[2];
  const float* qkv_w   = (const float*)d_in[3];
  const float* qkv_b   = (const float*)d_in[4];
  const float* proj_w  = (const float*)d_in[5];
  const float* proj_b  = (const float*)d_in[6];
  const float* rel_tab = (const float*)d_in[7];
  const float* n2w     = (const float*)d_in[8];
  const float* n2b     = (const float*)d_in[9];
  const float* fc1_w   = (const float*)d_in[10];
  const float* fc1_b   = (const float*)d_in[11];
  const float* fc2_w   = (const float*)d_in[12];
  const float* fc2_b   = (const float*)d_in[13];
  float* out = (float*)d_out;

  // workspace layout (256-byte aligned regions)
  char* wsb = (char*)d_ws;
  float*    x1   = (float*)(wsb + 0);               // 100352*128 f32 = 51,380,224 B
  _Float16* hbuf = (_Float16*)(wsb + 51380224);     // 100352*128 f16 = 25,690,112 B
  _Float16* wq_t = (_Float16*)(wsb + 77070336);     // 384*128 f16
  _Float16* pj_t = (_Float16*)(wsb + 77168640);     // 128*128 f16
  _Float16* f1_t = (_Float16*)(wsb + 77201408);     // 512*128 f16
  _Float16* f2_t = (_Float16*)(wsb + 77332480);     // 128*512 f16

  swin_wconv<<<256, 256, 0, stream>>>(qkv_w, proj_w, fc1_w, fc2_w,
                                      wq_t, pj_t, f1_t, f2_t);
  swin_attn<<<NWIN, 256, 0, stream>>>(x, n1w, n1b, wq_t, qkv_b, pj_t, proj_b,
                                      rel_tab, x1);
  swin_ln2<<<12544, 256, 0, stream>>>(x1, n2w, n2b, hbuf);
  swin_mlp<<<1568, 256, 0, stream>>>(x1, hbuf, f1_t, fc1_b, f2_t, fc2_b, out);
}
